// HybridDualStream_28346784153714
// MI455X (gfx1250) — matrix operand order repaired and checked by reading, not compiled, not run on hardware
//
#include <hip/hip_runtime.h>
#include <hip/hip_bf16.h>
#include <math.h>
#include <stdint.h>

// ---------------------------------------------------------------------------
// HybridDualStream for MI455X (gfx1250, wave32).
// All heavy GEMMs -> v_wmma_f32_16x16x32_bf16 (LDS-tiled, 128x128 block tile,
// 8 waves of 64x32). Global->LDS staging uses CDNA5 async-to-LDS when the
// toolchain exposes it (ASYNCcnt + s_wait_asynccnt), else sync ds_store path.
// Selective scan parallelized 16 lanes (=state dim) per (batch,channel) group.
// ---------------------------------------------------------------------------

typedef __attribute__((ext_vector_type(16))) __bf16 v16bf;
typedef __attribute__((ext_vector_type(8)))  __bf16 v8bf;
typedef __attribute__((ext_vector_type(8)))  float  v8f;
typedef int v4i_vs __attribute__((vector_size(16)));   // matches builtin param

#if defined(__has_builtin)
# if __has_builtin(__builtin_amdgcn_global_load_async_to_lds_b128)
#  define HAVE_ASYNC_LDS 1
# endif
#endif
#ifndef HAVE_ASYNC_LDS
# define HAVE_ASYNC_LDS 0
#endif

namespace {
constexpr int BK    = 16;          // batch
constexpr int LK    = 1024;        // sequence length
constexpr int DK    = 768;         // model dim
constexpr int DINK  = 1536;        // mamba inner dim
constexpr int SK    = 16;          // state dim
constexpr int RK    = 48;          // dt rank
constexpr int XDK   = 80;          // dt_rank + 2S
constexpr int NPROJ = 3072;        // 2*DIN
constexpr int MK    = BK * LK;     // 16384 tokens
constexpr float BN_SC = 0.99999500003749975f;  // 1/sqrt(1+1e-5)
}

#define DEV static __device__ __forceinline__

DEV float sigmoidf_(float x) { return 1.0f / (1.0f + __expf(-x)); }
DEV float siluf_(float x)    { return x * sigmoidf_(x); }
DEV float softplusf_(float x){ return x > 20.0f ? x : log1pf(__expf(x)); }
DEV float geluf_(float x)    { return 0.5f * x * (1.0f + erff(x * 0.70710678118654752f)); }

#if HAVE_ASYNC_LDS
// One 16-byte global->LDS async copy (global_load_async_to_lds_b128).
// Toolchain signature (from diagnostics): (v4i AS1*, v4i AS3*, imm, imm).
// Generic->AS1 is numerically identity; generic LDS low 32 bits are the LDS
// offset (aperture layout), so uintptr_t round-trips are valid.
DEV void async_copy_b128(const void* g, void* l)
{
    __builtin_amdgcn_global_load_async_to_lds_b128(
        (__attribute__((address_space(1))) v4i_vs*)(uintptr_t)g,
        (__attribute__((address_space(3))) v4i_vs*)(uintptr_t)l,
        0, 0);
}
DEV void wait_asynccnt0()
{
#if defined(__has_builtin) && __has_builtin(__builtin_amdgcn_s_wait_asynccnt)
    __builtin_amdgcn_s_wait_asynccnt(0);
#else
    asm volatile("s_wait_asynccnt 0x0" ::: "memory");
#endif
}
#endif

// ---------------------------------------------------------------------------
// fp32 -> bf16 conversion (weights + activations feeding WMMA)
// ---------------------------------------------------------------------------
__global__ __launch_bounds__(256)
void convert_bf16_kernel(const float* __restrict__ in, __bf16* __restrict__ out, size_t n)
{
    size_t i = (size_t)blockIdx.x * 256 + threadIdx.x;
    if (i < n) out[i] = (__bf16)in[i];
}

__global__ __launch_bounds__(256)
void aneg_kernel(const float* __restrict__ a_log, float* __restrict__ aneg, int n)
{
    int i = blockIdx.x * 256 + threadIdx.x;
    if (i < n) aneg[i] = -__expf(a_log[i]);
}

// ---------------------------------------------------------------------------
// LayerNorm over D=768; one block per token; optional f32 and bf16 outputs.
// ---------------------------------------------------------------------------
__global__ __launch_bounds__(256)
void layernorm_kernel(const float* __restrict__ x,
                      const float* __restrict__ g, const float* __restrict__ b,
                      float* __restrict__ out_f32, __bf16* __restrict__ out_bf)
{
    __shared__ float rs[256];
    __shared__ float rs2[256];
    const int m = blockIdx.x;
    const float* row = x + (size_t)m * DK;
    float v[3];
    float s = 0.0f, s2 = 0.0f;
    #pragma unroll
    for (int i = 0; i < 3; ++i) {
        v[i] = row[threadIdx.x + i * 256];
        s  += v[i];
        s2 += v[i] * v[i];
    }
    rs[threadIdx.x] = s; rs2[threadIdx.x] = s2;
    __syncthreads();
    for (int off = 128; off > 0; off >>= 1) {
        if (threadIdx.x < (unsigned)off) {
            rs[threadIdx.x]  += rs[threadIdx.x + off];
            rs2[threadIdx.x] += rs2[threadIdx.x + off];
        }
        __syncthreads();
    }
    const float mean = rs[0] * (1.0f / DK);
    const float var  = rs2[0] * (1.0f / DK) - mean * mean;
    const float inv  = rsqrtf(var + 1e-5f);
    #pragma unroll
    for (int i = 0; i < 3; ++i) {
        const int d = threadIdx.x + i * 256;
        const float o = (v[i] - mean) * inv * g[d] + b[d];
        if (out_f32) out_f32[(size_t)m * DK + d] = o;
        if (out_bf)  out_bf [(size_t)m * DK + d] = (__bf16)o;
    }
}

// ---------------------------------------------------------------------------
// Generic WMMA GEMM:  C[M,Nreal] = A[M,K](bf16) * W[Nreal,K]^T(bf16) (+bias)
// Block tile 128x128, K-step 32; wave tile 64x32 => 4x2 wmma per step.
// Fragment layouts used here (A and B share this k order):
//   A lane<16: row m, K k0..k0+15 (contiguous); lane>=16: K k0+16..k0+31.
//   B lane<16: col n, K k0..k0+15 (contiguous); lane>=16: K k0+16..k0+31.
// K must be a multiple of 32 and M a multiple of 128 (true for all calls).
// ---------------------------------------------------------------------------
__global__ __launch_bounds__(256)
void gemm_bf16_wmma(const __bf16* __restrict__ A,
                    const __bf16* __restrict__ W,
                    const float* __restrict__ bias,
                    float* __restrict__ C,
                    int M, int Nreal, int K)
{
    __shared__ __bf16 As[128][32];
    __shared__ __bf16 Bs[128][32];

    const int tid  = threadIdx.x;
    const int lane = tid & 31;
    const int wave = tid >> 5;
    const int wm   = (wave >> 2) * 64;   // wave M offset in tile: 0 or 64
    const int wn   = (wave & 3) * 32;    // wave N offset in tile: 0..96
    const int m0   = blockIdx.y * 128;
    const int n0   = blockIdx.x * 128;
    const int lr   = tid >> 1;           // 0..127: tile row this thread loads
    const int lk   = (tid & 1) * 16;     // 0 or 16: K-half this thread loads
    const int half = lane >> 4;
    const int l16  = lane & 15;

    v8f zacc = {};
    v8f acc[4][2];
    #pragma unroll
    for (int i = 0; i < 4; ++i)
        #pragma unroll
        for (int j = 0; j < 2; ++j) acc[i][j] = zacc;

    const int  wrow  = n0 + lr;
    const bool fullN = (n0 + 128 <= Nreal);  // block-uniform: tile fully in-range

    for (int k0 = 0; k0 < K; k0 += 32) {
#if HAVE_ASYNC_LDS
        if (k0) __syncthreads();             // prior compute done; LDS reusable
        const __bf16* ga = A + (size_t)(m0 + lr) * K + (k0 + lk);
        async_copy_b128(ga,     &As[lr][lk]);
        async_copy_b128(ga + 8, &As[lr][lk + 8]);
        if (fullN) {
            const __bf16* gb = W + (size_t)wrow * K + (k0 + lk);
            async_copy_b128(gb,     &Bs[lr][lk]);
            async_copy_b128(gb + 8, &Bs[lr][lk + 8]);
        } else {
            v16bf bv;
            #pragma unroll
            for (int q = 0; q < 16; ++q) bv[q] = (__bf16)0.0f;
            if (wrow < Nreal)
                bv = *(const v16bf*)(W + (size_t)wrow * K + (k0 + lk));
            *(v16bf*)(&Bs[lr][lk]) = bv;
        }
        wait_asynccnt0();                    // our async LDS writes landed
        __syncthreads();                     // everyone's data visible
#else
        v16bf av = *(const v16bf*)(A + (size_t)(m0 + lr) * K + (k0 + lk));
        v16bf bv;
        if (fullN) {
            bv = *(const v16bf*)(W + (size_t)wrow * K + (k0 + lk));
        } else {
            #pragma unroll
            for (int q = 0; q < 16; ++q) bv[q] = (__bf16)0.0f;
            if (wrow < Nreal)
                bv = *(const v16bf*)(W + (size_t)wrow * K + (k0 + lk));
        }
        __syncthreads();
        *(v16bf*)(&As[lr][lk]) = av;
        *(v16bf*)(&Bs[lr][lk]) = bv;
        __syncthreads();
#endif

        // ---- fragments ----
        v16bf afrag[4];
        #pragma unroll
        for (int i = 0; i < 4; ++i) {
            const __bf16* ap = &As[wm + i * 16 + l16][half * 16];
            v8bf lo = *(const v8bf*)(ap);
            v8bf hi = *(const v8bf*)(ap + 8);
            afrag[i] = __builtin_shufflevector(lo, hi,
                0, 1, 2, 3, 4, 5, 6, 7, 8, 9, 10, 11, 12, 13, 14, 15);
        }
        v16bf bfrag[2];
        #pragma unroll
        for (int j = 0; j < 2; ++j)
            bfrag[j] = *(const v16bf*)(&Bs[wn + j * 16 + l16][half * 16]);

        // ---- 8 WMMAs ----
        #pragma unroll
        for (int i = 0; i < 4; ++i)
            #pragma unroll
            for (int j = 0; j < 2; ++j)
                acc[i][j] = __builtin_amdgcn_wmma_f32_16x16x32_bf16(
                    false, afrag[i], false, bfrag[j],
                    (short)0, acc[i][j], false, false);
    }

    // ---- epilogue: C/D layout: vgpr r, lane<16: M=r, N=lane; lane>=16: M=8+r
    #pragma unroll
    for (int j = 0; j < 2; ++j) {
        const int col = n0 + wn + j * 16 + l16;
        if (col >= Nreal) continue;
        const float badd = bias ? bias[col] : 0.0f;
        #pragma unroll
        for (int i = 0; i < 4; ++i) {
            #pragma unroll
            for (int r = 0; r < 8; ++r) {
                const int row = m0 + wm + i * 16 + half * 8 + r;
                C[(size_t)row * Nreal + col] = acc[i][j][r] + badd;
            }
        }
    }
}

// ---------------------------------------------------------------------------
// Mamba causal depthwise conv (k=4) + SiLU over xs = xz[:, :DIN]
// ---------------------------------------------------------------------------
__global__ __launch_bounds__(256)
void mamba_conv_silu_kernel(const float* __restrict__ xz,
                            const float* __restrict__ cw,   // [DIN,4]
                            const float* __restrict__ cb,
                            float* __restrict__ uf, __bf16* __restrict__ ub)
{
    size_t idx = (size_t)blockIdx.x * 256 + threadIdx.x;
    if (idx >= (size_t)MK * DINK) return;
    const int    c = (int)(idx % DINK);
    const size_t m = idx / DINK;
    const int    l = (int)(m % LK);
    float a = cb[c];
    #pragma unroll
    for (int t = 0; t < 4; ++t) {
        const int lt = l + t - 3;
        if (lt >= 0)
            a += cw[c * 4 + t] * xz[(size_t)((long long)m + (t - 3)) * NPROJ + c];
    }
    const float s = siluf_(a);
    uf[idx] = s;
    ub[idx] = (__bf16)s;
}

// ---------------------------------------------------------------------------
// dt = softplus(dtp @ dt_w^T + dt_b); one block per token; dtp (48) in LDS.
// ---------------------------------------------------------------------------
__global__ __launch_bounds__(256)
void dt_kernel(const float* __restrict__ xdbl,   // [M,80]
               const float* __restrict__ dt_w,   // [DIN,48]
               const float* __restrict__ dt_b,
               float* __restrict__ dt)
{
    __shared__ float sp[RK];
    const int m = blockIdx.x;
    if (threadIdx.x < RK) sp[threadIdx.x] = xdbl[(size_t)m * XDK + threadIdx.x];
    __syncthreads();
    for (int c = threadIdx.x; c < DINK; c += 256) {
        float a = dt_b[c];
        const float* wr = dt_w + (size_t)c * RK;
        #pragma unroll 8
        for (int r = 0; r < RK; ++r) a += sp[r] * wr[r];
        dt[(size_t)m * DINK + c] = softplusf_(a);
    }
}

// ---------------------------------------------------------------------------
// Selective scan: 16 lanes (= state dim) per (b,c) group; h kept in a VGPR.
// y[m,c] = sum_s h_s * C_s + D[c]*u  after  h_s = exp(dt*A_s)*h_s + dt*B_s*u.
// ---------------------------------------------------------------------------
__global__ __launch_bounds__(256)
void scan_kernel(const float* __restrict__ u,     // [M,DIN]
                 const float* __restrict__ dt,    // [M,DIN]
                 const float* __restrict__ xdbl,  // [M,80]; B at 48, C at 64
                 const float* __restrict__ aneg,  // [DIN,16]
                 const float* __restrict__ dssm,  // [DIN]
                 float* __restrict__ y)           // [M,DIN]
{
    const int tid = threadIdx.x;
    const int grp = tid >> 4;                  // 16 groups per block
    const int s   = tid & 15;
    const int gid = blockIdx.x * 16 + grp;     // 0 .. B*DIN-1
    const int b   = gid / DINK;
    const int c   = gid - b * DINK;
    const float As = aneg[c * SK + s];
    const float Dc = dssm[c];
    float h = 0.0f;
    const size_t base = (size_t)b * LK;
    for (int l = 0; l < LK; ++l) {
        const size_t m = base + l;
        // prefetch next steps' streams (global_prefetch_b8)
        __builtin_prefetch(dt + (m + 4) * DINK + c, 0, 0);
        __builtin_prefetch(u  + (m + 4) * DINK + c, 0, 0);
        const float dtv = dt[m * DINK + c];       // broadcast across 16 lanes
        const float uv  = u [m * DINK + c];
        const float Bv  = xdbl[m * XDK + RK + s];
        const float Cv  = xdbl[m * XDK + RK + SK + s];
        const float dA  = __expf(dtv * As);
        h = dA * h + (dtv * Bv) * uv;
        float p = h * Cv;
        p += __shfl_xor(p, 1, 16);
        p += __shfl_xor(p, 2, 16);
        p += __shfl_xor(p, 4, 16);
        p += __shfl_xor(p, 8, 16);
        if (s == 0) y[m * DINK + c] = p + Dc * uv;
    }
}

// y * silu(z) -> bf16 (input to out_proj GEMM)
__global__ __launch_bounds__(256)
void gate_kernel(const float* __restrict__ y, const float* __restrict__ xz,
                 __bf16* __restrict__ out)
{
    size_t idx = (size_t)blockIdx.x * 256 + threadIdx.x;
    if (idx >= (size_t)MK * DINK) return;
    const int    c = (int)(idx % DINK);
    const size_t m = idx / DINK;
    const float  z = xz[m * NPROJ + DINK + c];
    out[idx] = (__bf16)(y[idx] * siluf_(z));
}

// id_filtered = id_seq + sigmoid(id_scale) * id_out
__global__ __launch_bounds__(256)
void id_final_kernel(const float* __restrict__ id_seq, const float* __restrict__ proj,
                     const float* __restrict__ id_scale, float* __restrict__ out)
{
    size_t idx = (size_t)blockIdx.x * 256 + threadIdx.x;
    if (idx >= (size_t)MK * DK) return;
    out[idx] = id_seq[idx] + sigmoidf_(id_scale[0]) * proj[idx];
}

// ---------------------------------------------------------------------------
// Attr branch: dilated depthwise conv + BN + GELU -> bf16 (pconv GEMM input)
// ---------------------------------------------------------------------------
__global__ __launch_bounds__(256)
void dwconv_bn_gelu_kernel(const float* __restrict__ x,     // at [M,768]
                           const float* __restrict__ w,     // [768,k]
                           const float* __restrict__ bias,
                           const float* __restrict__ g, const float* __restrict__ b,
                           __bf16* __restrict__ out, int k, int dil, int pad)
{
    size_t idx = (size_t)blockIdx.x * 256 + threadIdx.x;
    if (idx >= (size_t)MK * DK) return;
    const int    c = (int)(idx % DK);
    const size_t m = idx / DK;
    const int    l = (int)(m % LK);
    float a = bias[c];
    for (int t = 0; t < k; ++t) {
        const int lt = l + t * dil - pad;
        if (lt >= 0 && lt < LK)
            a += w[c * k + t] * x[(size_t)((long long)m + (t * dil - pad)) * DK + c];
    }
    const float tb = a * (g[c] * BN_SC) + b[c];
    out[idx] = (__bf16)geluf_(tb);
}

// BN(pconv output) -> bf16 slice of concat buffer [M, 3*768]
__global__ __launch_bounds__(256)
void bn_hcat_kernel(const float* __restrict__ p, const float* __restrict__ g,
                    const float* __restrict__ b, __bf16* __restrict__ hcat, int slice)
{
    size_t idx = (size_t)blockIdx.x * 256 + threadIdx.x;
    if (idx >= (size_t)MK * DK) return;
    const int    c = (int)(idx % DK);
    const size_t m = idx / DK;
    hcat[m * (size_t)(3 * DK) + (size_t)slice * DK + c] =
        (__bf16)(p[idx] * (g[c] * BN_SC) + b[c]);
}

// attr_filtered = at + sigmoid(res_w) * gelu(BN(fused))
__global__ __launch_bounds__(256)
void attr_final_kernel(const float* __restrict__ at, const float* __restrict__ fused,
                       const float* __restrict__ g, const float* __restrict__ b,
                       const float* __restrict__ res_w, float* __restrict__ out)
{
    size_t idx = (size_t)blockIdx.x * 256 + threadIdx.x;
    if (idx >= (size_t)MK * DK) return;
    const int c = (int)(idx % DK);
    const float t = fused[idx] * (g[c] * BN_SC) + b[c];
    out[idx] = at[idx] + sigmoidf_(res_w[0]) * geluf_(t);
}

// ---------------------------------------------------------------------------
// Host orchestration
// ---------------------------------------------------------------------------
extern "C" void kernel_launch(void* const* d_in, const int* in_sizes, int n_in,
                              void* d_out, int out_size, void* d_ws, size_t ws_size,
                              hipStream_t stream)
{
    const float* id_seq    = (const float*)d_in[0];
    const float* attr_seq  = (const float*)d_in[1];
    const float* ln_id_g   = (const float*)d_in[2];
    const float* ln_id_b   = (const float*)d_in[3];
    const float* ln_at_g   = (const float*)d_in[4];
    const float* ln_at_b   = (const float*)d_in[5];
    const float* in_proj_w = (const float*)d_in[6];
    const float* conv_w    = (const float*)d_in[7];
    const float* conv_b    = (const float*)d_in[8];
    const float* x_proj_w  = (const float*)d_in[9];
    const float* dt_w      = (const float*)d_in[10];
    const float* dt_b      = (const float*)d_in[11];
    const float* A_log     = (const float*)d_in[12];
    const float* D_ssm     = (const float*)d_in[13];
    const float* out_proj_w= (const float*)d_in[14];
    const float* id_scale  = (const float*)d_in[15];
    const float* fus_w     = (const float*)d_in[16];
    const float* fus_b     = (const float*)d_in[17];
    const float* gf        = (const float*)d_in[18];
    const float* bf_       = (const float*)d_in[19];
    const float* res_w     = (const float*)d_in[20];
    const float *dw_w[3], *dw_b[3], *ga[3], *ba[3], *pw_w[3], *pw_b[3], *gb[3], *bb[3];
    for (int br = 0; br < 3; ++br) {
        const int base = 21 + br * 8;
        dw_w[br] = (const float*)d_in[base + 0];
        dw_b[br] = (const float*)d_in[base + 1];
        ga[br]   = (const float*)d_in[base + 2];
        ba[br]   = (const float*)d_in[base + 3];
        pw_w[br] = (const float*)d_in[base + 4];
        pw_b[br] = (const float*)d_in[base + 5];
        gb[br]   = (const float*)d_in[base + 6];
        bb[br]   = (const float*)d_in[base + 7];
    }
    float* out_id   = (float*)d_out;                    // [M, 768]
    float* out_attr = (float*)d_out + (size_t)MK * DK;  // [M, 768]

    // ---- workspace bump allocator ----
    char* ws = (char*)d_ws;
    size_t off = 0;
    auto alloc = [&](size_t bytes) -> void* {
        off = (off + 255) & ~(size_t)255;
        void* p = ws + off;
        off += bytes;
        return p;
    };
    __bf16* wb_inproj  = (__bf16*)alloc((size_t)NPROJ * DK * 2);
    __bf16* wb_xproj   = (__bf16*)alloc((size_t)XDK * DINK * 2);
    __bf16* wb_outproj = (__bf16*)alloc((size_t)DK * DINK * 2);
    __bf16* wb_pw[3];
    for (int br = 0; br < 3; ++br) wb_pw[br] = (__bf16*)alloc((size_t)DK * DK * 2);
    __bf16* wb_fus     = (__bf16*)alloc((size_t)DK * 3 * DK * 2);
    float*  aneg       = (float*)alloc((size_t)DINK * SK * 4);
    __bf16* idn_bf     = (__bf16*)alloc((size_t)MK * DK * 2);
    float*  at_f32     = (float*)alloc((size_t)MK * DK * 4);
    float*  xz         = (float*)alloc((size_t)MK * NPROJ * 4);
    float*  u_f32      = (float*)alloc((size_t)MK * DINK * 4);
    __bf16* u_bf       = (__bf16*)alloc((size_t)MK * DINK * 2);
    float*  xdbl       = (float*)alloc((size_t)MK * XDK * 4);
    float*  dtbuf      = (float*)alloc((size_t)MK * DINK * 4);
    float*  ybuf       = (float*)alloc((size_t)MK * DINK * 4);
    __bf16* ygate_bf   = (__bf16*)alloc((size_t)MK * DINK * 2);
    float*  tmp768     = (float*)alloc((size_t)MK * DK * 4);   // reused GEMM out
    __bf16* branch_bf  = (__bf16*)alloc((size_t)MK * DK * 2);  // reused per branch
    __bf16* hcat_bf    = (__bf16*)alloc((size_t)MK * 3 * DK * 2);
    (void)ws_size; (void)in_sizes; (void)n_in; (void)out_size;

    auto cvt = [&](const float* src, __bf16* dst, size_t n) {
        convert_bf16_kernel<<<dim3((unsigned)((n + 255) / 256)), 256, 0, stream>>>(src, dst, n);
    };
    auto ew_grid = [](size_t n) { return dim3((unsigned)((n + 255) / 256)); };

    // ---- weight precision conversion (bf16 for WMMA) ----
    cvt(in_proj_w,  wb_inproj,  (size_t)NPROJ * DK);
    cvt(x_proj_w,   wb_xproj,   (size_t)XDK * DINK);
    cvt(out_proj_w, wb_outproj, (size_t)DK * DINK);
    for (int br = 0; br < 3; ++br) cvt(pw_w[br], wb_pw[br], (size_t)DK * DK);
    cvt(fus_w, wb_fus, (size_t)DK * 3 * DK);
    aneg_kernel<<<dim3((DINK * SK + 255) / 256), 256, 0, stream>>>(A_log, aneg, DINK * SK);

    // ==================== ID stream (Mamba) ====================
    layernorm_kernel<<<dim3(MK), 256, 0, stream>>>(id_seq, ln_id_g, ln_id_b,
                                                   (float*)nullptr, idn_bf);
    // xz = LN(id) @ in_proj_w^T : [16384,3072]
    gemm_bf16_wmma<<<dim3(NPROJ / 128, MK / 128), 256, 0, stream>>>(
        idn_bf, wb_inproj, nullptr, xz, MK, NPROJ, DK);
    // causal depthwise conv(k=4) + SiLU
    mamba_conv_silu_kernel<<<ew_grid((size_t)MK * DINK), 256, 0, stream>>>(
        xz, conv_w, conv_b, u_f32, u_bf);
    // xdbl = xs @ x_proj_w^T : [16384,80]  (N=80 guarded inside GEMM)
    gemm_bf16_wmma<<<dim3(1, MK / 128), 256, 0, stream>>>(
        u_bf, wb_xproj, nullptr, xdbl, MK, XDK, DINK);
    // dt = softplus(dtp @ dt_w^T + dt_b)
    dt_kernel<<<dim3(MK), 256, 0, stream>>>(xdbl, dt_w, dt_b, dtbuf);
    // selective scan: 16 lanes/channel, 24576 groups
    scan_kernel<<<dim3(BK * DINK / 16), 256, 0, stream>>>(
        u_f32, dtbuf, xdbl, aneg, D_ssm, ybuf);
    // gate with silu(z)
    gate_kernel<<<ew_grid((size_t)MK * DINK), 256, 0, stream>>>(ybuf, xz, ygate_bf);
    // id_out = (y*silu(z)) @ out_proj_w^T
    gemm_bf16_wmma<<<dim3(DK / 128, MK / 128), 256, 0, stream>>>(
        ygate_bf, wb_outproj, nullptr, tmp768, MK, DK, DINK);
    id_final_kernel<<<ew_grid((size_t)MK * DK), 256, 0, stream>>>(
        id_seq, tmp768, id_scale, out_id);

    // ==================== Attr stream (multi-scale CNN) ====================
    layernorm_kernel<<<dim3(MK), 256, 0, stream>>>(attr_seq, ln_at_g, ln_at_b,
                                                   at_f32, (__bf16*)nullptr);
    const int kk[3]  = {3, 5, 7};
    const int dil[3] = {1, 2, 3};
    const int pad[3] = {1, 4, 9};
    for (int br = 0; br < 3; ++br) {
        dwconv_bn_gelu_kernel<<<ew_grid((size_t)MK * DK), 256, 0, stream>>>(
            at_f32, dw_w[br], dw_b[br], ga[br], ba[br], branch_bf,
            kk[br], dil[br], pad[br]);
        gemm_bf16_wmma<<<dim3(DK / 128, MK / 128), 256, 0, stream>>>(
            branch_bf, wb_pw[br], pw_b[br], tmp768, MK, DK, DK);
        bn_hcat_kernel<<<ew_grid((size_t)MK * DK), 256, 0, stream>>>(
            tmp768, gb[br], bb[br], hcat_bf, br);
    }
    // fused = concat(f1,f2,f3) @ fus_w^T + fus_b  (K=2304)
    gemm_bf16_wmma<<<dim3(DK / 128, MK / 128), 256, 0, stream>>>(
        hcat_bf, wb_fus, fus_b, tmp768, MK, DK, 3 * DK);
    attr_final_kernel<<<ew_grid((size_t)MK * DK), 256, 0, stream>>>(
        at_f32, tmp768, gf, bf_, res_w, out_attr);
}
